// SwinTransformer_6004364280662
// MI455X (gfx1250) — matrix operand order repaired and checked by reading, not compiled, not run on hardware
//
#include <hip/hip_runtime.h>
#include <hip/hip_bf16.h>

typedef __attribute__((ext_vector_type(16))) __bf16 v16bf;
typedef __attribute__((ext_vector_type(2)))  __bf16 v2bf;
typedef __attribute__((ext_vector_type(8)))  float  v8f;

#define WSZ   7
#define SFT   3
#define NHD   3
#define CDIM  96
#define HIMG  224
#define WIMG  224
#define NTOK  49
#define NPAD  64
#define QKSCL 0.17677669529663687f   // 32^-0.5

#if defined(__AMDGCN__) && __has_builtin(__builtin_amdgcn_global_load_async_to_lds_b128)
#define HAVE_ASYNC 1
// exact param types per hipcc diagnostic: vector_size(16) int, AS1 (global) / AS3 (LDS)
typedef int v4i __attribute__((vector_size(16)));
typedef __attribute__((address_space(1))) v4i gas_v4i;
typedef __attribute__((address_space(3))) v4i las_v4i;
#else
#define HAVE_ASYNC 0
#endif

#if defined(__AMDGCN__) && __has_builtin(__builtin_amdgcn_cvt_pk_bf16_f32)
#define HAVE_CVTPK 1
#else
#define HAVE_CVTPK 0
#endif

// ---------- bf16 helpers ----------
__device__ __forceinline__ unsigned short f2bf_bits(float f) {
  unsigned u = __builtin_bit_cast(unsigned, f);
  u += 0x7FFFu + ((u >> 16) & 1u);            // round-to-nearest-even
  return (unsigned short)(u >> 16);
}
__device__ __forceinline__ __bf16 f2bf(float f) {
  unsigned short h = f2bf_bits(f);
  return __builtin_bit_cast(__bf16, h);
}
// packed 2xf32 -> 2xbf16: single V_CVT_PK_BF16_F32 (builtin if declared, else inline asm)
__device__ __forceinline__ v2bf f2bf2(float a, float b) {
#if HAVE_CVTPK
  return __builtin_amdgcn_cvt_pk_bf16_f32(a, b);
#elif defined(__AMDGCN__)
  unsigned r;
  asm("v_cvt_pk_bf16_f32 %0, %1, %2" : "=v"(r) : "v"(a), "v"(b));
  return __builtin_bit_cast(v2bf, r);
#else
  v2bf r; r[0] = f2bf(a); r[1] = f2bf(b); return r;
#endif
}
__device__ __forceinline__ v8f v8f_zero() {
  v8f z;
#pragma unroll
  for (int i = 0; i < 8; ++i) z[i] = 0.f;
  return z;
}

// ---------- vectorized WMMA fragment loaders (CDNA5 wave32, ISA 7.12.2) ----------
__device__ __forceinline__ v16bf frag_a_v(const __bf16* __restrict__ base, int ld, int lane) {
  int m = lane & 15, g = lane >> 4;
  const __bf16* p = base + m * ld + 16 * g;
  union { uint4 u[2]; v16bf v; } c;
  c.u[0] = *(const uint4*)(p);        // f[0..7]  : k = 16g .. 16g+7
  c.u[1] = *(const uint4*)(p + 8);    // f[8..15] : k = 16g+8 .. 16g+15
  return c.v;
}
__device__ __forceinline__ v16bf frag_bt_v(const __bf16* __restrict__ base, int ld, int lane) {
  int n = lane & 15, g = lane >> 4;
  const __bf16* p = base + n * ld + 16 * g;
  union { uint4 u[2]; v16bf v; } c;
  c.u[0] = *(const uint4*)(p);        // f[0..7]  : k = 16g .. 16g+7
  c.u[1] = *(const uint4*)(p + 8);    // f[8..15] : k = 16g+8 .. 16g+15
  return c.v;
}
// pre-swizzled global weight panel: one 32B contiguous load per lane
__device__ __forceinline__ v16bf frag_sw(const __bf16* __restrict__ panel, int tile, int lane) {
  const uint4* p = (const uint4*)(panel + tile * 512 + lane * 16);
  union { uint4 u[2]; v16bf v; } c;
  c.u[0] = p[0]; c.u[1] = p[1];
  return c.v;
}
__device__ __forceinline__ v8f wmma_bf(v16bf a, v16bf b, v8f c) {
  return __builtin_amdgcn_wmma_f32_16x16x32_bf16(false, a, false, b, (short)0, c, false, false);
}

// ================= Kernel P: fp32 weights -> bf16 fragment-swizzled panels =================
__global__ void __launch_bounds__(256) swin_prep_w(
    const float* __restrict__ qkv_w, const float* __restrict__ proj_w,
    const float* __restrict__ fc1_w, const float* __restrict__ fc2_w,
    __bf16* __restrict__ Wqkv, __bf16* __restrict__ Wp,
    __bf16* __restrict__ W1, __bf16* __restrict__ W2) {
  int i = blockIdx.x * 256 + threadIdx.x;
  int lane = (i >> 4) & 31, e = i & 15;
  int kh = 16 * (lane >> 4) + e, nl = lane & 15;
  if (i < 27648) {                               // QKV panel: K=96, N=288, ntiles=18
    int tile = i >> 9;
    int kt = tile / 18, nt = tile - kt * 18;
    int k = kt * 32 + kh, n = nt * 16 + nl;
    float w = qkv_w[n * 96 + k];
    if (n < 96) w *= QKSCL;                      // fold q-scale into weights
    Wqkv[i] = f2bf(w);
  } else if (i < 27648 + 9216) {                 // proj panel: K=96, N=96, ntiles=6
    int j = i - 27648; int t2 = j >> 9;
    int kt = t2 / 6, nt = t2 - kt * 6;
    Wp[j] = f2bf(proj_w[(nt * 16 + nl) * 96 + kt * 32 + kh]);
  } else if (i < 27648 + 9216 + 36864) {         // fc1 panel: K=96, N=384, ntiles=24
    int j = i - 27648 - 9216; int t2 = j >> 9;
    int kt = t2 / 24, nt = t2 - kt * 24;
    W1[j] = f2bf(fc1_w[(nt * 16 + nl) * 96 + kt * 32 + kh]);
  } else if (i < 27648 + 9216 + 36864 + 36864) { // fc2 panel: K=384, N=96, ntiles=6
    int j = i - 27648 - 9216 - 36864; int t2 = j >> 9;
    int kt = t2 / 6, nt = t2 - kt * 6;
    W2[j] = f2bf(fc2_w[(nt * 16 + nl) * 384 + kt * 32 + kh]);
  }
}

// ============ Kernel A: LN1 + cyclic shift + window partition -> bf16 xw ============
__global__ void __launch_bounds__(256) swin_ln1_part(
    const float* __restrict__ x, const float* __restrict__ g, const float* __restrict__ bta,
    __bf16* __restrict__ xw) {
  const int wg = blockIdx.x;
  const int lane = threadIdx.x & 31, wv = threadIdx.x >> 5;
  const int b = wg >> 10, win = wg & 1023;
  const int wr = win >> 5, wc = win & 31;
  __bf16* dst = xw + (size_t)wg * NPAD * CDIM;
  for (int n = wv; n < NPAD; n += 8) {
    if (n < NTOK) {
      int r = n / 7, c = n - r * 7;
      int hh = wr * 7 + r + SFT; if (hh >= HIMG) hh -= HIMG;
      int ww = wc * 7 + c + SFT; if (ww >= WIMG) ww -= WIMG;
      const float* row = x + ((size_t)b * (HIMG * WIMG) + hh * WIMG + ww) * CDIM;
      float v0 = row[lane], v1 = row[lane + 32], v2 = row[lane + 64];
      float s1 = v0 + v1 + v2;
      float s2 = v0 * v0 + v1 * v1 + v2 * v2;
#pragma unroll
      for (int off = 16; off; off >>= 1) {
        s1 += __shfl_xor(s1, off, 32);
        s2 += __shfl_xor(s2, off, 32);
      }
      float mu = s1 * (1.f / 96.f);
      float var = s2 * (1.f / 96.f) - mu * mu;
      float rs = rsqrtf(var + 1e-5f);
      dst[n * CDIM + lane]      = f2bf((v0 - mu) * rs * g[lane]      + bta[lane]);
      dst[n * CDIM + lane + 32] = f2bf((v1 - mu) * rs * g[lane + 32] + bta[lane + 32]);
      dst[n * CDIM + lane + 64] = f2bf((v2 - mu) * rs * g[lane + 64] + bta[lane + 64]);
    } else {
      __bf16 z = f2bf(0.f);
      dst[n * CDIM + lane] = z; dst[n * CDIM + lane + 32] = z; dst[n * CDIM + lane + 64] = z;
    }
  }
}

// ===== Kernel B: fused window attention (QKV -> S -> softmax -> PV -> proj + residual) =====
__global__ void __launch_bounds__(256) swin_win_attn(
    const __bf16* __restrict__ xw, const __bf16* __restrict__ Wqkv,
    const float* __restrict__ qkv_b, const float* __restrict__ rpb,
    const float* __restrict__ mask, const __bf16* __restrict__ Wp,
    const float* __restrict__ proj_b, const float* __restrict__ xin,
    float* __restrict__ out) {
  __shared__ __align__(16) unsigned char smem[61440];
  __bf16* sQK  = (__bf16*)smem;                 // [64][192] bf16: Q(0..95) | K(96..191)
  __bf16* sVT  = (__bf16*)(smem + 24576);       // [96][64]  bf16: V transposed (d-major)
  __bf16* sX   = (__bf16*)(smem + 36864);       // [64][96]  bf16  (phase 1)
  float*  sS   = (float*) (smem + 36864);       // [64][64]  f32: S^T, sS[j*64+i] (per head)
  __bf16* sO   = (__bf16*)(smem + 36864);       // [64][32]  bf16  (reuses sS region)
  __bf16* sP   = (__bf16*)(smem + 53248);       // [64][64]  bf16 (row-major P)

  const int wg = blockIdx.x;
  const int lane = threadIdx.x & 31, wv = threadIdx.x >> 5;
  const int col = lane & 15, grp = lane >> 4;
  const int b = wg >> 10, win = wg & 1023;
  const int wr = win >> 5, wc = win & 31;

  if (threadIdx.x < 216) __builtin_prefetch(Wqkv + threadIdx.x * 128, 0, 1);
  if (threadIdx.x < 72)  __builtin_prefetch(Wp + threadIdx.x * 128, 0, 1);

  // stage activation tile (64x96 bf16 = 12KB) into LDS (async, tracked by ASYNCcnt)
  {
    const __bf16* gsrc = xw + (size_t)wg * NPAD * CDIM;
#if HAVE_ASYNC
#pragma unroll
    for (int it = 0; it < 3; ++it) {
      int off = (it * 256 + threadIdx.x) * 16;
      __builtin_amdgcn_global_load_async_to_lds_b128(
          (gas_v4i*)((const char*)gsrc + off),
          (las_v4i*)((char*)sX + off), 0, 0);
    }
#if __has_builtin(__builtin_amdgcn_s_wait_asynccnt)
    __builtin_amdgcn_s_wait_asynccnt(0);
#else
    asm volatile("s_wait_asynccnt 0" ::: "memory");
#endif
#else
    const uint4* s4 = (const uint4*)gsrc;
    uint4* d4 = (uint4*)sX;
    for (int i = threadIdx.x; i < 768; i += 256) d4[i] = s4[i];
#endif
  }
  __syncthreads();

  // ---- Phase 1: QKV = X @ Wqkv (M=64,N=288,K=96). Q,K row-major; V stored transposed ----
  for (int t = wv; t < 72; t += 8) {
    int mt = t & 3, nt = t >> 2;
    v8f acc = v8f_zero();
#pragma unroll
    for (int kk = 0; kk < 3; ++kk) {
      v16bf a  = frag_a_v(sX + mt * 16 * CDIM + kk * 32, CDIM, lane);
      v16bf bb = frag_sw(Wqkv, kk * 18 + nt, lane);
      acc = wmma_bf(a, bb, acc);
    }
    int n = nt * 16 + col;
    float bias = qkv_b[n];
    if (nt < 6) bias *= QKSCL;                  // q slice (wave-uniform)
    if (nt < 12) {                              // Q,K -> row-major (wave-uniform branch)
#pragma unroll
      for (int jp = 0; jp < 4; ++jp) {
        v2bf r = f2bf2(acc[2 * jp] + bias, acc[2 * jp + 1] + bias);
        int m = mt * 16 + 2 * jp + 8 * grp;
        sQK[m * 192 + n]       = r[0];
        sQK[(m + 1) * 192 + n] = r[1];
      }
    } else {                                    // V -> transposed, contiguous 16B store
      int d = n - 192;
      union { v2bf h[4]; uint4 u; } pk;
#pragma unroll
      for (int jp = 0; jp < 4; ++jp)
        pk.h[jp] = f2bf2(acc[2 * jp] + bias, acc[2 * jp + 1] + bias);
      *(uint4*)&sVT[d * 64 + mt * 16 + 8 * grp] = pk.u;
    }
  }
  __syncthreads();

  // ---- Phase 2: per-head attention, proj accumulated in registers ----
  v8f zacc[3];
#pragma unroll
  for (int i = 0; i < 3; ++i) zacc[i] = v8f_zero();

  for (int h = 0; h < NHD; ++h) {
    // S = Qh @ Kh^T (64x64, K=32); store S TRANSPOSED: sS[j*64+i] -> contiguous lane stores
    for (int t = wv; t < 16; t += 8) {
      int mt = t & 3, nt = t >> 2;
      v16bf a  = frag_a_v (sQK + mt * 16 * 192 + h * 32, 192, lane);
      v16bf bb = frag_bt_v(sQK + nt * 16 * 192 + 96 + h * 32, 192, lane);
      v8f acc = wmma_bf(a, bb, v8f_zero());
      float* dstc = &sS[(nt * 16 + col) * 64 + mt * 16 + 8 * grp];
      *(float4*)(dstc)     = make_float4(acc[0], acc[1], acc[2], acc[3]);
      *(float4*)(dstc + 4) = make_float4(acc[4], acc[5], acc[6], acc[7]);
    }
    __syncthreads();

    // softmax per row i (S^T layout: column j strided by 64)
    if (threadIdx.x < 64) {
      int i = threadIdx.x;
      unsigned* sPu = (unsigned*)(sP + i * 64);
      if (i < NTOK) {
        int r1 = i / 7, c1 = i - r1 * 7;
        const float* mrow = mask + ((size_t)win * NTOK + i) * NTOK;
        float mx = -1e30f;
        for (int j = 0; j < NTOK; ++j) {
          int r2 = j / 7, c2 = j - r2 * 7;
          int ridx = (r1 - r2 + 6) * 13 + (c1 - c2 + 6);
          float v = sS[j * 64 + i] + rpb[ridx * NHD + h] + mrow[j];
          sS[j * 64 + i] = v;
          mx = fmaxf(mx, v);
        }
        float sum = 0.f;
        for (int j = 0; j < NTOK; ++j) {
          float e = __expf(sS[j * 64 + i] - mx);
          sS[j * 64 + i] = e;
          sum += e;
        }
        float inv = 1.f / sum;
        for (int jj = 0; jj < 24; ++jj) {   // 48 values packed in pairs
          v2bf r = f2bf2(sS[(2 * jj) * 64 + i] * inv, sS[(2 * jj + 1) * 64 + i] * inv);
          sPu[jj] = __builtin_bit_cast(unsigned, r);
        }
        {                                    // j = 48 valid, 49..63 zero
          v2bf r = f2bf2(sS[48 * 64 + i] * inv, 0.f);
          sPu[24] = __builtin_bit_cast(unsigned, r);
        }
        for (int jj = 25; jj < 32; ++jj) sPu[jj] = 0u;
      } else {
        for (int jj = 0; jj < 32; ++jj) sPu[jj] = 0u;
      }
    }
    __syncthreads();   // sP ready, sS dead -> sO may overwrite that region

    // O = P @ Vh (64x32, K=64): B from transposed V
    {
      int mt = wv & 3, nt = wv >> 2;
      v8f acc = v8f_zero();
#pragma unroll
      for (int kk = 0; kk < 2; ++kk) {
        v16bf a  = frag_a_v (sP + mt * 16 * 64 + kk * 32, 64, lane);
        v16bf bb = frag_bt_v(sVT + (h * 32 + nt * 16) * 64 + kk * 32, 64, lane);
        acc = wmma_bf(a, bb, acc);
      }
#pragma unroll
      for (int jp = 0; jp < 4; ++jp) {
        v2bf r = f2bf2(acc[2 * jp], acc[2 * jp + 1]);
        int m = mt * 16 + 2 * jp + 8 * grp;
        sO[m * 32 + nt * 16 + col]       = r[0];
        sO[(m + 1) * 32 + nt * 16 + col] = r[1];
      }
    }
    __syncthreads();

    // Z += Oh @ Wp[h-slice] (64x96, K=32)
#pragma unroll
    for (int idx = 0; idx < 3; ++idx) {
      int t = wv + idx * 8;
      int mt = t & 3, nt = t >> 2;
      v16bf a  = frag_a_v(sO + mt * 16 * 32, 32, lane);
      v16bf bb = frag_sw(Wp, h * 6 + nt, lane);
      zacc[idx] = wmma_bf(a, bb, zacc[idx]);
    }
    __syncthreads();   // proj reads done before next head's S overwrites sO region
  }

  // ---- Phase 3: reverse shift + residual, write x1 = x + attn(x) ----
#pragma unroll
  for (int idx = 0; idx < 3; ++idx) {
    int t = wv + idx * 8;
    int mt = t & 3, nt = t >> 2;
    int ch = nt * 16 + col;
    float pb = proj_b[ch];
#pragma unroll
    for (int j = 0; j < 8; ++j) {
      int n = mt * 16 + j + 8 * grp;
      if (n < NTOK) {
        int r = n / 7, c = n - r * 7;
        int hh = wr * 7 + r + SFT; if (hh >= HIMG) hh -= HIMG;
        int ww = wc * 7 + c + SFT; if (ww >= WIMG) ww -= WIMG;
        size_t off = ((size_t)b * (HIMG * WIMG) + hh * WIMG + ww) * CDIM + ch;
        out[off] = xin[off] + zacc[idx][j] + pb;
      }
    }
  }
}

// ======= Kernel C: fused MLP, in-place: out = out + FC2(GELU(FC1(LN2(out)))) =======
__global__ void __launch_bounds__(256) swin_mlp(
    float* __restrict__ out, const __bf16* __restrict__ W1, const float* __restrict__ b1,
    const __bf16* __restrict__ W2, const float* __restrict__ b2,
    const float* __restrict__ g2, const float* __restrict__ bt2) {
  __shared__ __align__(16) unsigned char smem[61440];
  __bf16* sM = (__bf16*)smem;             // [64][96]  LN2 output
  __bf16* sT = (__bf16*)(smem + 12288);   // [64][384] GELU(FC1)
  const int lane = threadIdx.x & 31, wv = threadIdx.x >> 5;
  const int col = lane & 15, grp = lane >> 4;
  const size_t r0 = (size_t)blockIdx.x * 64;

  if (threadIdx.x < 288) __builtin_prefetch(W1 + (threadIdx.x & 255) * 128, 0, 1);

  // LN2 (one token row per wave iteration)
  for (int n = wv; n < 64; n += 8) {
    const float* row = out + (r0 + n) * CDIM;
    float v0 = row[lane], v1 = row[lane + 32], v2 = row[lane + 64];
    float s1 = v0 + v1 + v2;
    float s2 = v0 * v0 + v1 * v1 + v2 * v2;
#pragma unroll
    for (int off = 16; off; off >>= 1) {
      s1 += __shfl_xor(s1, off, 32);
      s2 += __shfl_xor(s2, off, 32);
    }
    float mu = s1 * (1.f / 96.f);
    float var = s2 * (1.f / 96.f) - mu * mu;
    float rs = rsqrtf(var + 1e-5f);
    sM[n * CDIM + lane]      = f2bf((v0 - mu) * rs * g2[lane]      + bt2[lane]);
    sM[n * CDIM + lane + 32] = f2bf((v1 - mu) * rs * g2[lane + 32] + bt2[lane + 32]);
    sM[n * CDIM + lane + 64] = f2bf((v2 - mu) * rs * g2[lane + 64] + bt2[lane + 64]);
  }
  __syncthreads();

  // FC1 + exact GELU (M=64, N=384, K=96)
  for (int t = wv; t < 96; t += 8) {
    int mt = t & 3, nt = t >> 2;
    v8f acc = v8f_zero();
#pragma unroll
    for (int kk = 0; kk < 3; ++kk) {
      v16bf a  = frag_a_v(sM + mt * 16 * CDIM + kk * 32, CDIM, lane);
      v16bf bb = frag_sw(W1, kk * 24 + nt, lane);
      acc = wmma_bf(a, bb, acc);
    }
    int n = nt * 16 + col;
    float bb1 = b1[n];
#pragma unroll
    for (int jp = 0; jp < 4; ++jp) {
      float u0 = acc[2 * jp] + bb1, u1 = acc[2 * jp + 1] + bb1;
      float g0 = 0.5f * u0 * (1.f + erff(u0 * 0.70710678f));
      float g1 = 0.5f * u1 * (1.f + erff(u1 * 0.70710678f));
      v2bf r = f2bf2(g0, g1);
      int m = mt * 16 + 2 * jp + 8 * grp;
      sT[m * 384 + n]       = r[0];
      sT[(m + 1) * 384 + n] = r[1];
    }
  }
  __syncthreads();

  // FC2 + residual (M=64, N=96, K=384)
  for (int t = wv; t < 24; t += 8) {
    int mt = t & 3, nt = t >> 2;
    v8f acc = v8f_zero();
#pragma unroll
    for (int kk = 0; kk < 12; ++kk) {
      v16bf a  = frag_a_v(sT + mt * 16 * 384 + kk * 32, 384, lane);
      v16bf bb = frag_sw(W2, kk * 6 + nt, lane);
      acc = wmma_bf(a, bb, acc);
    }
    int ch = nt * 16 + col;
    float bb2 = b2[ch];
#pragma unroll
    for (int j = 0; j < 8; ++j) {
      size_t row = r0 + mt * 16 + j + 8 * grp;
      out[row * CDIM + ch] += acc[j] + bb2;   // rows exclusive to this block
    }
  }
}

// ================================ host launcher ================================
extern "C" void kernel_launch(void* const* d_in, const int* in_sizes, int n_in,
                              void* d_out, int out_size, void* d_ws, size_t ws_size,
                              hipStream_t stream) {
  const float* x      = (const float*)d_in[0];
  const float* mask   = (const float*)d_in[1];
  const float* n1g    = (const float*)d_in[2];
  const float* n1b    = (const float*)d_in[3];
  const float* qkv_w  = (const float*)d_in[4];
  const float* qkv_b  = (const float*)d_in[5];
  const float* rpb    = (const float*)d_in[6];
  const float* proj_w = (const float*)d_in[7];
  const float* proj_b = (const float*)d_in[8];
  const float* n2g    = (const float*)d_in[9];
  const float* n2b    = (const float*)d_in[10];
  const float* fc1_w  = (const float*)d_in[11];
  const float* fc1_b  = (const float*)d_in[12];
  const float* fc2_w  = (const float*)d_in[13];
  const float* fc2_b  = (const float*)d_in[14];
  float* out = (float*)d_out;

  char* ws = (char*)d_ws;
  const size_t OFF_XW   = 0;                          // 8192*64*96*2 = 100,663,296
  const size_t OFF_WQKV = 100663296;                  // 27648*2
  const size_t OFF_WP   = OFF_WQKV + 55296;           // 9216*2
  const size_t OFF_W1   = OFF_WP   + 18432;           // 36864*2
  const size_t OFF_W2   = OFF_W1   + 73728;           // 36864*2
  __bf16* xw   = (__bf16*)(ws + OFF_XW);
  __bf16* Wqkv = (__bf16*)(ws + OFF_WQKV);
  __bf16* Wp   = (__bf16*)(ws + OFF_WP);
  __bf16* W1   = (__bf16*)(ws + OFF_W1);
  __bf16* W2   = (__bf16*)(ws + OFF_W2);

  hipLaunchKernelGGL(swin_prep_w, dim3(432), dim3(256), 0, stream,
                     qkv_w, proj_w, fc1_w, fc2_w, Wqkv, Wp, W1, W2);
  hipLaunchKernelGGL(swin_ln1_part, dim3(8192), dim3(256), 0, stream,
                     x, n1g, n1b, xw);
  hipLaunchKernelGGL(swin_win_attn, dim3(8192), dim3(256), 0, stream,
                     xw, Wqkv, qkv_b, rpb, mask, Wp, proj_b, x, out);
  hipLaunchKernelGGL(swin_mlp, dim3(6272), dim3(256), 0, stream,
                     out, W1, fc1_b, W2, fc2_b, n2g, n2b);
}
